// OrnamentProcessor_2929167695969
// MI455X (gfx1250) — matrix operand order repaired and checked by reading, not compiled, not run on hardware
//
#include <hip/hip_runtime.h>
#include <stdint.h>

#define BQ 16
#define SQ 2048
#define DQ 512
#define HQ 256
#define MROWS (BQ*SQ)   // 32768

typedef __bf16 bf16;
typedef __attribute__((ext_vector_type(16))) bf16  v16bf;
typedef __attribute__((ext_vector_type(8)))  bf16  v8bf;
typedef __attribute__((ext_vector_type(8)))  float v8f;
typedef __attribute__((ext_vector_type(4)))  unsigned int u32x4;
typedef __attribute__((ext_vector_type(8)))  int  i32x8;
typedef __attribute__((ext_vector_type(4)))  int  i32x4;

// ---------------- TDM: 2D bf16 tile -> LDS (D# per CDNA5 ISA §8) ----------------
__device__ __forceinline__ void tdm_load_2d_bf16(uint32_t lds_byte_addr,
                                                 const bf16* gsrc,
                                                 uint32_t stride_elems,
                                                 uint32_t tile_d0, uint32_t tile_d1)
{
    uint64_t ga = (uint64_t)(uintptr_t)gsrc;
    u32x4 g0;
    g0[0] = 1u;                                   // count=1, user mode, no gather
    g0[1] = lds_byte_addr;                        // lds_addr
    g0[2] = (uint32_t)ga;                         // global_addr[31:0]
    g0[3] = (uint32_t)((ga >> 32) & 0x01FFFFFFu)  // global_addr[56:32]
          | (2u << 30);                           // type=2 (image)
    const uint32_t td0 = stride_elems;            // tensor_dim0 (elements)
    const uint32_t td1 = 1u << 24;                // generous row bound
    i32x8 g1;
    g1[0] = (int)(1u << 16);                      // data_size=1 -> 2 bytes
    g1[1] = (int)((td0 & 0xFFFFu) << 16);         // tensor_dim0[15:0]
    g1[2] = (int)((td0 >> 16) | ((td1 & 0xFFFFu) << 16));
    g1[3] = (int)((td1 >> 16) | (tile_d0 << 16)); // tile_dim0
    g1[4] = (int)(tile_d1 & 0xFFFFu);             // tile_dim1 (tile_dim2=0)
    g1[5] = (int)stride_elems;                    // tensor_dim0_stride[31:0]
    g1[6] = 0;
    g1[7] = 0;
    i32x4 gz4 = {0, 0, 0, 0};
    i32x8 gz8 = {0, 0, 0, 0, 0, 0, 0, 0};
    // 6-arg variant (clang-23 / therock-10.0 headers)
    __builtin_amdgcn_tensor_load_to_lds(g0, g1, gz4, gz4, gz8, 0);
}

__device__ __forceinline__ v16bf cat8(v8bf lo, v8bf hi)
{
    v16bf r;
#pragma unroll
    for (int i = 0; i < 8; ++i) { r[i] = lo[i]; r[i + 8] = hi[i]; }
    return r;
}

__device__ __forceinline__ v8f wmma_bf16(v16bf a, v16bf b, v8f c)
{
    return __builtin_amdgcn_wmma_f32_16x16x32_bf16(false, a, false, b,
                                                   (short)0, c, false, false);
}

// ---------------- Generic WMMA GEMM:  C[M,N] = A[M,K] @ W[N,K]^T ----------------
// block = 256 threads (8 waves), tile 128x64, BK=32, TDM double-buffered LDS.
template <int OUT_BF16>
__global__ __launch_bounds__(256)
void gemm_wmma_kernel(const bf16* __restrict__ A, int lda,
                      const bf16* __restrict__ W, int ldw,
                      void* __restrict__ C, int ldc, int KT)
{
    __shared__ bf16 Abuf[2][128 * 32];
    __shared__ bf16 Bbuf[2][64 * 32];
    const int tid = threadIdx.x, lane = tid & 31, wid = tid >> 5;
    const int wm = wid & 3, wn = wid >> 2;          // 4x2 wave grid -> 32x32 per wave
    const int row0 = blockIdx.x * 128, col0 = blockIdx.y * 64;

    v8f acc[2][2];
#pragma unroll
    for (int i = 0; i < 2; i++)
#pragma unroll
        for (int j = 0; j < 2; j++) acc[i][j] = 0.f;

    if (wid == 0) {
        tdm_load_2d_bf16((uint32_t)(uintptr_t)&Abuf[0][0], A + (size_t)row0 * lda, lda, 32, 128);
        tdm_load_2d_bf16((uint32_t)(uintptr_t)&Bbuf[0][0], W + (size_t)col0 * ldw, ldw, 32, 64);
        __builtin_amdgcn_s_wait_tensorcnt(0);
    }
    __syncthreads();

    for (int kt = 0; kt < KT; ++kt) {
        const int cur = kt & 1;
        if (wid == 0 && (kt + 1) < KT) {
            tdm_load_2d_bf16((uint32_t)(uintptr_t)&Abuf[cur ^ 1][0],
                             A + (size_t)row0 * lda + (kt + 1) * 32, lda, 32, 128);
            tdm_load_2d_bf16((uint32_t)(uintptr_t)&Bbuf[cur ^ 1][0],
                             W + (size_t)col0 * ldw + (kt + 1) * 32, ldw, 32, 64);
        }
        const bf16* Ab = &Abuf[cur][0];
        const bf16* Bb = &Bbuf[cur][0];
        const int c0 = (lane >> 4) * 16;
        v16bf afrag[2], bfrag[2];
#pragma unroll
        for (int i = 0; i < 2; i++) {           // A 16x32 fragment (contiguous k = 16*h + s)
            const int r = wm * 32 + i * 16 + (lane & 15);
            afrag[i] = cat8(*(const v8bf*)(Ab + r * 32 + c0),
                            *(const v8bf*)(Ab + r * 32 + c0 + 8));
        }
#pragma unroll
        for (int j = 0; j < 2; j++) {           // B 32x16 fragment: lane=N, elems=K
            const int r = wn * 32 + j * 16 + (lane & 15);
            bfrag[j] = *(const v16bf*)(Bb + r * 32 + (lane >> 4) * 16);
        }
#pragma unroll
        for (int i = 0; i < 2; i++)
#pragma unroll
            for (int j = 0; j < 2; j++)
                acc[i][j] = wmma_bf16(afrag[i], bfrag[j], acc[i][j]);

        if (wid == 0) __builtin_amdgcn_s_wait_tensorcnt(0);
        __syncthreads();
    }

#pragma unroll
    for (int i = 0; i < 2; i++)
#pragma unroll
        for (int j = 0; j < 2; j++) {
            const int mbase = row0 + wm * 32 + i * 16 + ((lane >> 4) * 8);
            const int n = col0 + wn * 32 + j * 16 + (lane & 15);
#pragma unroll
            for (int r = 0; r < 8; r++) {
                const size_t off = (size_t)(mbase + r) * ldc + n;
                if (OUT_BF16) ((bf16*)C)[off] = (bf16)acc[i][j][r];
                else          ((float*)C)[off] = acc[i][j][r];
            }
        }
}

// ---------------- LSTM recurrent scan (one block per direction) ----------------
// z_t = Zx[t] + h_{t-1} @ Whh^T + b ; gates -> c,h. WMMA: M=16 batch, N=1024, K=256.
// nt loop deliberately NOT unrolled: keeps Whh B-fragments streaming (no LICM ->
// no 512-VGPR hoist -> no scratch spills); Whh stays L2-resident (2 MB total).
__global__ __launch_bounds__(256)
void lstm_scan_kernel(const bf16* __restrict__ Zx,   // [32768, 2048] (b*S+t rows)
                      const bf16* __restrict__ Whh,  // [2048, 256] (dir-major)
                      const float* __restrict__ bias,// [2048]
                      bf16* __restrict__ hout, int ostride, int coff_base)
{
    __shared__ bf16  hbuf[16 * HQ];      // h, bf16, batch-major
    __shared__ float cbuf[16 * HQ];      // c, f32
    __shared__ bf16  zbuf[16 * 1024];    // gate pre-activations
    const int dir = blockIdx.x;
    const int tid = threadIdx.x, lane = tid & 31, wid = tid >> 5;
    const int coff = coff_base + dir * HQ;
    const bf16*  Wd = Whh + (size_t)dir * 1024 * HQ;
    const float* bd = bias + dir * 1024;

    for (int i = tid; i < 16 * HQ; i += 256) { hbuf[i] = (bf16)0.f; cbuf[i] = 0.f; }
    __syncthreads();

    for (int step = 0; step < SQ; ++step) {
        const int t = dir ? (SQ - 1 - step) : step;
        // ---- recurrent GEMM phase ----
        const int c0 = (lane >> 4) * 16;
        const int rowm = lane & 15;                 // batch row
        v16bf afrag[8];
#pragma unroll
        for (int kt = 0; kt < 8; kt++) {
            const int base = rowm * HQ + kt * 32 + c0;
            afrag[kt] = cat8(*(const v8bf*)(hbuf + base),
                             *(const v8bf*)(hbuf + base + 8));
        }
#pragma unroll 1
        for (int nt = 0; nt < 8; ++nt) {            // 8 waves x 8 tiles = 1024 cols
            const int n0 = (wid * 8 + nt) * 16;
            v8f accv = 0.f;
#pragma unroll
            for (int kt = 0; kt < 8; kt++) {
                const bf16* wp = Wd + (size_t)(n0 + (lane & 15)) * HQ
                               + kt * 32 + (lane >> 4) * 16;
                accv = wmma_bf16(afrag[kt], *(const v16bf*)wp, accv);
            }
            const int ncol = n0 + (lane & 15);
            const float bval = bd[ncol];
#pragma unroll
            for (int r = 0; r < 8; r++) {
                const int m = r + ((lane >> 4) * 8);  // batch
                const float z = accv[r] + bval +
                    (float)Zx[((size_t)m * SQ + t) * 2048 + dir * 1024 + ncol];
                zbuf[m * 1024 + ncol] = (bf16)z;
            }
        }
        __syncthreads();
        // ---- gate mix / state update: 16 contiguous units per thread, vectorized ----
        {
            const int b = tid >> 4;
            const int j0 = (tid & 15) * 16;
            if (step + 1 < SQ) {
                const int tn = dir ? (SQ - 2 - step) : (step + 1);
                __builtin_prefetch(&Zx[((size_t)b * SQ + tn) * 2048 + dir * 1024 + j0 * 4], 0, 1);
            }
            const int zb = b * 1024 + j0;
#pragma unroll
            for (int h8 = 0; h8 < 2; h8++) {
                const v8bf vi = *(const v8bf*)&zbuf[zb +       h8 * 8];
                const v8bf vf = *(const v8bf*)&zbuf[zb + 256 + h8 * 8];
                const v8bf vg = *(const v8bf*)&zbuf[zb + 512 + h8 * 8];
                const v8bf vo = *(const v8bf*)&zbuf[zb + 768 + h8 * 8];
                v8bf hv;
#pragma unroll
                for (int e = 0; e < 8; e++) {
                    const float ig = 1.f / (1.f + __expf(-(float)vi[e]));
                    const float fg = 1.f / (1.f + __expf(-(float)vf[e]));
                    const float og = 1.f / (1.f + __expf(-(float)vo[e]));
                    const float gg = tanhf((float)vg[e]);
                    const int ci = b * HQ + j0 + h8 * 8 + e;
                    const float c = fg * cbuf[ci] + ig * gg;
                    cbuf[ci] = c;
                    hv[e] = (bf16)(og * tanhf(c));
                }
                *(v8bf*)&hbuf[b * HQ + j0 + h8 * 8] = hv;
                *(v8bf*)&hout[((size_t)b * SQ + t) * ostride + coff + j0 + h8 * 8] = hv;
            }
        }
        __syncthreads();
    }
}

// ---------------- LayerNorm(features) -> f32 + bf16 into cat[:, :512] ----------
__global__ __launch_bounds__(256)
void ln_cast_kernel(const float* __restrict__ x, const float* __restrict__ g,
                    const float* __restrict__ b, float* __restrict__ y,
                    bf16* __restrict__ catb)
{
    __shared__ float s_sum[256], s_sq[256];
    const size_t row = blockIdx.x;
    const int tid = threadIdx.x;
    const float v0 = x[row * DQ + tid];
    const float v1 = x[row * DQ + 256 + tid];
    s_sum[tid] = v0 + v1;
    s_sq[tid]  = v0 * v0 + v1 * v1;
    __syncthreads();
    for (int s2 = 128; s2 > 0; s2 >>= 1) {
        if (tid < s2) { s_sum[tid] += s_sum[tid + s2]; s_sq[tid] += s_sq[tid + s2]; }
        __syncthreads();
    }
    const float mean = s_sum[0] * (1.f / DQ);
    const float var  = s_sq[0] * (1.f / DQ) - mean * mean;
    const float rstd = rsqrtf(var + 1e-5f);
    const float o0 = (v0 - mean) * rstd * g[tid] + b[tid];
    const float o1 = (v1 - mean) * rstd * g[256 + tid] + b[256 + tid];
    y[row * DQ + tid] = o0;
    y[row * DQ + 256 + tid] = o1;
    catb[row * 1024 + tid] = (bf16)o0;
    catb[row * 1024 + 256 + tid] = (bf16)o1;
}

// ---------------- relu(LN(z + gb1)) -> bf16 (row length 1024) ------------------
__global__ __launch_bounds__(256)
void ln_relu_kernel(const float* __restrict__ z, const float* __restrict__ gb1,
                    const float* __restrict__ g, const float* __restrict__ b,
                    bf16* __restrict__ out)
{
    __shared__ float s_sum[256], s_sq[256];
    const size_t row = blockIdx.x;
    const int tid = threadIdx.x;
    float v[4], sum = 0.f, sq = 0.f;
#pragma unroll
    for (int i = 0; i < 4; i++) {
        const int col = tid + i * 256;
        v[i] = z[row * 1024 + col] + gb1[col];
        sum += v[i]; sq += v[i] * v[i];
    }
    s_sum[tid] = sum; s_sq[tid] = sq;
    __syncthreads();
    for (int s2 = 128; s2 > 0; s2 >>= 1) {
        if (tid < s2) { s_sum[tid] += s_sum[tid + s2]; s_sq[tid] += s_sq[tid + s2]; }
        __syncthreads();
    }
    const float mean = s_sum[0] * (1.f / 1024.f);
    const float var  = s_sq[0] * (1.f / 1024.f) - mean * mean;
    const float rstd = rsqrtf(var + 1e-5f);
#pragma unroll
    for (int i = 0; i < 4; i++) {
        const int col = tid + i * 256;
        const float o = (v[i] - mean) * rstd * g[col] + b[col];
        out[row * 1024 + col] = (bf16)fmaxf(o, 0.f);
    }
}

// ---------------- sequential decide scan (lane = batch) ------------------------
__global__ void decide_kernel(const float* __restrict__ rand,
                              const unsigned char* __restrict__ mask,
                              int* __restrict__ apply)
{
    const int b = threadIdx.x;
    if (b >= BQ) return;
    int last = -1000000000, cons = 0;
    for (int s2 = 0; s2 < SQ; ++s2) {
        const float r = rand[b * SQ + s2];
        const int m = mask[b * SQ + s2] != 0;
        const int iok = (s2 == 0) || (s2 - last >= 2);
        const int ap = m && (r < 0.5f) && iok && (cons < 2);
        if (ap) { last = s2; cons = cons + 1; }
        else if (m) { cons = 0; }
        apply[b * SQ + s2] = ap;
    }
}

// ---------------- epilogue: bias + range-wise transform + masked select --------
__global__ __launch_bounds__(256)
void final_kernel(float* __restrict__ out, const float* __restrict__ lnx,
                  const float* __restrict__ gb2, const int* __restrict__ apply)
{
    const size_t idx = (size_t)blockIdx.x * 256 + threadIdx.x;
    const int d = (int)(idx & (DQ - 1));
    const size_t row = idx >> 9;
    const float t = out[idx] + gb2[d];
    float proc;
    if      (d < 256) proc = fminf(fmaxf(t, 0.f), 1.f); // clip01 (P)
    else if (d < 384) proc = t * 0.3f;                  // DUR_F
    else if (d < 448) proc = t * 0.7f;                  // VEL_F
    else              proc = t + 0.05f;                 // POS_SHIFT
    out[idx] = apply[row] ? proc : lnx[idx];
}

__global__ __launch_bounds__(256)
void cast_kernel(const float* __restrict__ src, bf16* __restrict__ dst, int n)
{
    const int i = blockIdx.x * 256 + threadIdx.x;
    if (i < n) dst[i] = (bf16)src[i];
}

extern "C" void kernel_launch(void* const* d_in, const int* in_sizes, int n_in,
                              void* d_out, int out_size, void* d_ws, size_t ws_size,
                              hipStream_t stream)
{
    (void)in_sizes; (void)n_in; (void)out_size; (void)ws_size;
    const float* features = (const float*)d_in[0];
    const unsigned char* amask = (const unsigned char*)d_in[1];
    const float* rand_vals = (const float*)d_in[2];
    const float* ln_g = (const float*)d_in[3];
    const float* ln_b = (const float*)d_in[4];
    const float* Wih0 = (const float*)d_in[5];
    const float* Whh0 = (const float*)d_in[6];
    const float* b0   = (const float*)d_in[7];
    const float* Wih1 = (const float*)d_in[8];
    const float* Whh1 = (const float*)d_in[9];
    const float* b1   = (const float*)d_in[10];
    const float* gW1  = (const float*)d_in[11];
    const float* gb1  = (const float*)d_in[12];
    const float* gln_g= (const float*)d_in[13];
    const float* gln_b= (const float*)d_in[14];
    const float* gW2  = (const float*)d_in[15];
    const float* gb2  = (const float*)d_in[16];
    float* outp = (float*)d_out;

    char* ws = (char*)d_ws;
    float* LNX  = (float*)(ws + 0);            // 64 MB  LN(x) f32
    bf16*  CAT  = (bf16*)(ws + 67108864);      // 64 MB  cat bf16 (later reused as g)
    char*  ZREG = ws + 134217728;              // 128 MB Z bf16 / MLP1 f32 (reused)
    bf16*  Zbf  = (bf16*)ZREG;
    float* Zf32 = (float*)ZREG;
    bf16*  H1   = (bf16*)(ws + 268435456);     // 32 MB  layer1 input bf16
    bf16*  WP   = (bf16*)(ws + 301989888);     // 9 MB   bf16 weights
    bf16* wih0b = WP;
    bf16* whh0b = wih0b + 1048576;
    bf16* wih1b = whh0b + 524288;
    bf16* whh1b = wih1b + 1048576;
    bf16* gw1b  = whh1b + 524288;
    bf16* gw2b  = gw1b + 1048576;              // only first 512 rows of gW2
    int*  APPLY = (int*)(ws + 311427072);

    // weights -> bf16
    cast_kernel<<<4096, 256, 0, stream>>>(Wih0, wih0b, 1048576);
    cast_kernel<<<2048, 256, 0, stream>>>(Whh0, whh0b, 524288);
    cast_kernel<<<4096, 256, 0, stream>>>(Wih1, wih1b, 1048576);
    cast_kernel<<<2048, 256, 0, stream>>>(Whh1, whh1b, 524288);
    cast_kernel<<<4096, 256, 0, stream>>>(gW1, gw1b, 1048576);
    cast_kernel<<<2048, 256, 0, stream>>>(gW2, gw2b, 524288);

    // LN + bf16 copy into cat[:, :512]
    ln_cast_kernel<<<MROWS, 256, 0, stream>>>(features, ln_g, ln_b, LNX, CAT);

    // layer0: Z0 = LN(x) @ Wih0^T (both dirs, N=2048), then scan
    gemm_wmma_kernel<1><<<dim3(256, 32), 256, 0, stream>>>(CAT, 1024, wih0b, 512, Zbf, 2048, 16);
    lstm_scan_kernel<<<2, 256, 0, stream>>>(Zbf, whh0b, b0, H1, 512, 0);
    // layer1: Z1 = h1 @ Wih1^T, then scan (writes into cat[:, 512:])
    gemm_wmma_kernel<1><<<dim3(256, 32), 256, 0, stream>>>(H1, 512, wih1b, 512, Zbf, 2048, 16);
    lstm_scan_kernel<<<2, 256, 0, stream>>>(Zbf, whh1b, b1, CAT, 1024, 512);
    // MLP1: cat @ gW1^T -> f32 (reuse Z region), then relu(LN(.)) -> g (reuse CAT)
    gemm_wmma_kernel<0><<<dim3(256, 16), 256, 0, stream>>>(CAT, 1024, gw1b, 1024, Zf32, 1024, 32);
    ln_relu_kernel<<<MROWS, 256, 0, stream>>>(Zf32, gb1, gln_g, gln_b, CAT);
    // MLP2 (only first 512 output features needed): g @ gW2[:512]^T -> d_out
    gemm_wmma_kernel<0><<<dim3(256, 8), 256, 0, stream>>>(CAT, 1024, gw2b, 1024, outp, 512, 32);

    decide_kernel<<<1, 32, 0, stream>>>(rand_vals, amask, APPLY);
    final_kernel<<<(MROWS * DQ) / 256, 256, 0, stream>>>(outp, LNX, gb2, APPLY);
}
